// Generator_20787641712795
// MI455X (gfx1250) — matrix operand order repaired and checked by reading, not compiled, not run on hardware
//
#include <hip/hip_runtime.h>
#include <hip/hip_bf16.h>

typedef __bf16 bf16_t;
typedef bf16_t v16bf __attribute__((ext_vector_type(16)));
typedef bf16_t v8bf  __attribute__((ext_vector_type(8)));
typedef float  v8f   __attribute__((ext_vector_type(8)));

#define B_SZ 256
#define T_SZ 256

__device__ __forceinline__ float sigmoidf_(float x) {
    return 1.0f / (1.0f + __expf(-x));
}

// A-fragment (16x32 bf16, MxK). p is lane-resolved: base + row*stride + lh*8.
// Lanes 0-15: K=[0..7],[16..23]; lanes 16-31: K=[8..15],[24..31].
__device__ __forceinline__ v16bf ld_a(const bf16_t* __restrict__ p) {
    v8bf lo = *(const v8bf*)p;
    v8bf hi = *(const v8bf*)(p + 16);
    v16bf a;
#pragma unroll
    for (int i = 0; i < 8; ++i) { a[i] = lo[i]; a[i + 8] = hi[i]; }
    return a;
}

// B-fragment (32x16 bf16, KxN) from pre-transposed BT[N,K] row-major.
// p is lane-resolved: base + (n0+lm)*stride + lh*8. K=[8lh..8lh+7],[16+8lh..16+8lh+7].
__device__ __forceinline__ v16bf ld_b(const bf16_t* __restrict__ p) {
    v8bf lo = *(const v8bf*)p;
    v8bf hi = *(const v8bf*)(p + 16);
    v16bf b;
#pragma unroll
    for (int i = 0; i < 8; ++i) { b[i] = lo[i]; b[i + 8] = hi[i]; }
    return b;
}

#define WMMA_BF16(A, Bm, C) \
    __builtin_amdgcn_wmma_f32_16x16x32_bf16(false, (A), false, (Bm), (short)0, (C), false, false)

// One fused GRU timestep (reset_after semantics):
//   z = sig(x@Wz + h@Rz + bz), r = sig(x@Wr + h@Rr + br),
//   hh = act(x@Wh + b0h + r*(h@Rh + b1h)), h' = z*h + (1-z)*hh
// 128 threads = 4 waves; wave w owns cols [n0+w*16, n0+w*16+16) and 32 rows
// (two 16-row M-tiles -> every B fragment is reused by 2 WMMAs).
__global__ void __launch_bounds__(128)
gru_step_kernel(const bf16_t* __restrict__ xseq,   // [B, T, Din] bf16
                int Din, int t,
                const bf16_t* __restrict__ hin_bf, // [B, u] bf16
                const float*  __restrict__ hin_f,  // [B, u] f32
                const bf16_t* __restrict__ WT,     // [3u, Din] bf16 (W transposed)
                const bf16_t* __restrict__ RT,     // [3u, u]   bf16 (R transposed)
                const float*  __restrict__ b0,     // [3u]
                const float*  __restrict__ b1,     // [3u]
                int u,
                float*  __restrict__ hout_f,       // [B, u]
                bf16_t* __restrict__ hout_bf,      // [B, u]
                bf16_t* __restrict__ seq_out,      // [B, T, u] bf16 or null
                float*  __restrict__ final_out,    // [B, T, u] f32 or null
                int act_sigmoid)
{
    const int lane = threadIdx.x & 31;
    const int wave = threadIdx.x >> 5;           // 0..3 -> 16-col slice
    const int m0 = blockIdx.y * 32;              // two 16-row tiles: m0, m0+16
    const int n0 = blockIdx.x * 64 + wave * 16;
    const int lm = lane & 15;
    const int lh = (lane >> 4) & 1;

    v8f az0 = {}, az1 = {}, ar0 = {}, ar1 = {};
    v8f axh0 = {}, axh1 = {}, arh0 = {}, arh1 = {};

    // ---- lane-resolved stream pointers ----
    const long xstride = (long)T_SZ * Din;
    const bf16_t* ax0 = xseq + (long)(m0 + lm) * xstride + (long)t * Din + lh * 8;
    const bf16_t* ax1 = ax0 + 16 * xstride;

    const bf16_t* ah0 = hin_bf + (long)(m0 + lm) * u + lh * 8;
    const bf16_t* ah1 = ah0 + 16 * u;

    const long nrow = n0 + lm;
    const bf16_t* pwz = WT + ((long)(0 * u) + nrow) * Din + lh * 8;
    const bf16_t* pwr = WT + ((long)(1 * u) + nrow) * Din + lh * 8;
    const bf16_t* pwh = WT + ((long)(2 * u) + nrow) * Din + lh * 8;
    const bf16_t* prz = RT + ((long)(0 * u) + nrow) * u + lh * 8;
    const bf16_t* prr = RT + ((long)(1 * u) + nrow) * u + lh * 8;
    const bf16_t* prh = RT + ((long)(2 * u) + nrow) * u + lh * 8;

    // ================= input projection: K over Din (pipelined) =================
    {
        v16bf a0c = ld_a(ax0), a1c = ld_a(ax1);
        v16bf bzc = ld_b(pwz), brc = ld_b(pwr), bhc = ld_b(pwh);
        for (int k0 = 32; k0 < Din; k0 += 32) {
            // prefetch weight stream one chunk ahead into near caches
            __builtin_prefetch(pwz + k0 + 32, 0, 3);
            __builtin_prefetch(pwr + k0 + 32, 0, 3);
            __builtin_prefetch(pwh + k0 + 32, 0, 3);
            // issue next-iteration loads before consuming current fragments
            v16bf a0n = ld_a(ax0 + k0);
            v16bf a1n = ld_a(ax1 + k0);
            v16bf bzn = ld_b(pwz + k0);
            v16bf brn = ld_b(pwr + k0);
            v16bf bhn = ld_b(pwh + k0);
            az0  = WMMA_BF16(a0c, bzc, az0);
            az1  = WMMA_BF16(a1c, bzc, az1);
            ar0  = WMMA_BF16(a0c, brc, ar0);
            ar1  = WMMA_BF16(a1c, brc, ar1);
            axh0 = WMMA_BF16(a0c, bhc, axh0);
            axh1 = WMMA_BF16(a1c, bhc, axh1);
            a0c = a0n; a1c = a1n; bzc = bzn; brc = brn; bhc = bhn;
        }
        az0  = WMMA_BF16(a0c, bzc, az0);
        az1  = WMMA_BF16(a1c, bzc, az1);
        ar0  = WMMA_BF16(a0c, brc, ar0);
        ar1  = WMMA_BF16(a1c, brc, ar1);
        axh0 = WMMA_BF16(a0c, bhc, axh0);
        axh1 = WMMA_BF16(a1c, bhc, axh1);
    }

    // ================= recurrent part: K over u (pipelined) =================
    {
        v16bf a0c = ld_a(ah0), a1c = ld_a(ah1);
        v16bf bzc = ld_b(prz), brc = ld_b(prr), bhc = ld_b(prh);
        for (int k0 = 32; k0 < u; k0 += 32) {
            __builtin_prefetch(prz + k0 + 32, 0, 3);
            __builtin_prefetch(prr + k0 + 32, 0, 3);
            __builtin_prefetch(prh + k0 + 32, 0, 3);
            v16bf a0n = ld_a(ah0 + k0);
            v16bf a1n = ld_a(ah1 + k0);
            v16bf bzn = ld_b(prz + k0);
            v16bf brn = ld_b(prr + k0);
            v16bf bhn = ld_b(prh + k0);
            az0  = WMMA_BF16(a0c, bzc, az0);
            az1  = WMMA_BF16(a1c, bzc, az1);
            ar0  = WMMA_BF16(a0c, brc, ar0);
            ar1  = WMMA_BF16(a1c, brc, ar1);
            arh0 = WMMA_BF16(a0c, bhc, arh0);
            arh1 = WMMA_BF16(a1c, bhc, arh1);
            a0c = a0n; a1c = a1n; bzc = bzn; brc = brn; bhc = bhn;
        }
        az0  = WMMA_BF16(a0c, bzc, az0);
        az1  = WMMA_BF16(a1c, bzc, az1);
        ar0  = WMMA_BF16(a0c, brc, ar0);
        ar1  = WMMA_BF16(a1c, brc, ar1);
        arh0 = WMMA_BF16(a0c, bhc, arh0);
        arh1 = WMMA_BF16(a1c, bhc, arh1);
    }

    // ================= gate math on f32 accumulators =================
    // C/D layout: element g of v8f, lane L -> M = mtile + g + 8*(L>=16), N = n0+(L&15)
    const int nc = n0 + lm;
    const float bzc = b0[nc]         + b1[nc];
    const float brc = b0[u + nc]     + b1[u + nc];
    const float bh0 = b0[2 * u + nc];
    const float bh1 = b1[2 * u + nc];

    const v8f azv[2]  = { az0, az1 };
    const v8f arv[2]  = { ar0, ar1 };
    const v8f axhv[2] = { axh0, axh1 };
    const v8f arhv[2] = { arh0, arh1 };

#pragma unroll
    for (int mt = 0; mt < 2; ++mt) {
        const int mbase = m0 + mt * 16 + 8 * lh;
#pragma unroll
        for (int g = 0; g < 8; ++g) {
            const int m = mbase + g;
            const float z = sigmoidf_(azv[mt][g] + bzc);
            const float r = sigmoidf_(arv[mt][g] + brc);
            float cand = axhv[mt][g] + bh0 + r * (arhv[mt][g] + bh1);
            cand = act_sigmoid ? sigmoidf_(cand) : fmaxf(cand, 0.0f);
            const float hold = hin_f[(long)m * u + nc];
            const float hn = z * hold + (1.0f - z) * cand;
            hout_f[(long)m * u + nc]  = hn;
            hout_bf[(long)m * u + nc] = (bf16_t)hn;
            if (seq_out)   seq_out[((long)m * T_SZ + t) * u + nc]   = (bf16_t)hn;
            if (final_out) final_out[((long)m * T_SZ + t) * u + nc] = hn;
        }
    }
}

// fp32 [K,N] -> bf16 [N,K] (transpose + downconvert), run once per launch.
__global__ void transpose_cast_kernel(const float* __restrict__ src,
                                      bf16_t* __restrict__ dst, int K, int N) {
    long i = (long)blockIdx.x * blockDim.x + threadIdx.x;
    long total = (long)K * N;
    if (i >= total) return;
    int k = (int)(i / N);
    int n = (int)(i % N);
    dst[(long)n * K + k] = (bf16_t)src[i];
}

__global__ void cast_kernel(const float* __restrict__ src,
                            bf16_t* __restrict__ dst, long n) {
    long i = (long)blockIdx.x * blockDim.x + threadIdx.x;
    if (i < n) dst[i] = (bf16_t)src[i];
}

__global__ void zero_state_kernel(float* __restrict__ hf,
                                  bf16_t* __restrict__ hb, long n) {
    long i = (long)blockIdx.x * blockDim.x + threadIdx.x;
    if (i < n) { hf[i] = 0.0f; hb[i] = (bf16_t)0.0f; }
}

extern "C" void kernel_launch(void* const* d_in, const int* in_sizes, int n_in,
                              void* d_out, int out_size, void* d_ws, size_t ws_size,
                              hipStream_t stream) {
    const float* x  = (const float*)d_in[0];
    const float* W1 = (const float*)d_in[1];
    const float* R1 = (const float*)d_in[2];
    const float* b1 = (const float*)d_in[3];
    const float* W2 = (const float*)d_in[4];
    const float* R2 = (const float*)d_in[5];
    const float* b2 = (const float*)d_in[6];
    const float* W3 = (const float*)d_in[7];
    const float* R3 = (const float*)d_in[8];
    const float* b3 = (const float*)d_in[9];
    (void)in_sizes; (void)n_in; (void)out_size; (void)ws_size;

    const int Bv = 256, Tv = 256, D = 128, U1 = 512, U3 = 64;

    char* ws = (char*)d_ws;
    size_t off = 0;
    auto carve = [&](size_t bytes) -> char* {
        char* p = ws + off;
        off += (bytes + 255) & ~(size_t)255;
        return p;
    };
    bf16_t* WT1 = (bf16_t*)carve((size_t)3 * U1 * D  * 2);
    bf16_t* RT1 = (bf16_t*)carve((size_t)3 * U1 * U1 * 2);
    bf16_t* WT2 = (bf16_t*)carve((size_t)3 * U1 * U1 * 2);
    bf16_t* RT2 = (bf16_t*)carve((size_t)3 * U1 * U1 * 2);
    bf16_t* WT3 = (bf16_t*)carve((size_t)3 * U3 * U1 * 2);
    bf16_t* RT3 = (bf16_t*)carve((size_t)3 * U3 * U3 * 2);
    bf16_t* xbf  = (bf16_t*)carve((size_t)Bv * Tv * D  * 2);
    bf16_t* seq1 = (bf16_t*)carve((size_t)Bv * Tv * U1 * 2);
    bf16_t* seq2 = (bf16_t*)carve((size_t)Bv * Tv * U1 * 2);
    float*  hf0  = (float*) carve((size_t)Bv * U1 * 4);
    float*  hf1  = (float*) carve((size_t)Bv * U1 * 4);
    bf16_t* hb0  = (bf16_t*)carve((size_t)Bv * U1 * 2);
    bf16_t* hb1  = (bf16_t*)carve((size_t)Bv * U1 * 2);

    float*  hfs[2] = { hf0, hf1 };
    bf16_t* hbs[2] = { hb0, hb1 };

    auto tconv = [&](const float* s, bf16_t* d, int K, int N) {
        long tot = (long)K * N;
        int blocks = (int)((tot + 255) / 256);
        transpose_cast_kernel<<<blocks, 256, 0, stream>>>(s, d, K, N);
    };
    tconv(W1, WT1, D,  3 * U1);
    tconv(R1, RT1, U1, 3 * U1);
    tconv(W2, WT2, U1, 3 * U1);
    tconv(R2, RT2, U1, 3 * U1);
    tconv(W3, WT3, U1, 3 * U3);
    tconv(R3, RT3, U3, 3 * U3);

    {
        long n = (long)Bv * Tv * D;
        cast_kernel<<<(int)((n + 255) / 256), 256, 0, stream>>>(x, xbf, n);
    }

    // ---------------- layer 1: D=128 -> U1=512, relu candidate ----------------
    {
        long n = (long)Bv * U1;
        zero_state_kernel<<<(int)((n + 255) / 256), 256, 0, stream>>>(hf0, hb0, n);
        dim3 grid(U1 / 64, Bv / 32);
        for (int t = 0; t < Tv; ++t) {
            int rd = t & 1, wr = (t + 1) & 1;
            gru_step_kernel<<<grid, 128, 0, stream>>>(
                xbf, D, t, hbs[rd], hfs[rd], WT1, RT1, b1, b1 + 3 * U1, U1,
                hfs[wr], hbs[wr], seq1, nullptr, 0);
        }
    }

    // ---------------- layer 2: U1 -> U1, relu candidate ----------------
    {
        long n = (long)Bv * U1;
        zero_state_kernel<<<(int)((n + 255) / 256), 256, 0, stream>>>(hf0, hb0, n);
        dim3 grid(U1 / 64, Bv / 32);
        for (int t = 0; t < Tv; ++t) {
            int rd = t & 1, wr = (t + 1) & 1;
            gru_step_kernel<<<grid, 128, 0, stream>>>(
                seq1, U1, t, hbs[rd], hfs[rd], WT2, RT2, b2, b2 + 3 * U1, U1,
                hfs[wr], hbs[wr], seq2, nullptr, 0);
        }
    }

    // ---------------- layer 3: U1 -> U3=64, sigmoid candidate, f32 out ----------------
    {
        long n = (long)Bv * U3;
        zero_state_kernel<<<(int)((n + 255) / 256), 256, 0, stream>>>(hf0, hb0, n);
        dim3 grid(U3 / 64, Bv / 32);
        for (int t = 0; t < Tv; ++t) {
            int rd = t & 1, wr = (t + 1) & 1;
            gru_step_kernel<<<grid, 128, 0, stream>>>(
                seq2, U1, t, hbs[rd], hfs[rd], WT3, RT3, b3, b3 + 3 * U3, U3,
                hfs[wr], hbs[wr], nullptr, (float*)d_out, 1);
        }
    }
}
